// NR_GraphAttention_30219389894759
// MI455X (gfx1250) — compile-verified, matrix operand order checked by reading, not run on hardware
//
#include <hip/hip_runtime.h>
#include <hip/hip_bf16.h>

#define NN 50000
#define EE 400000
#define RR 1000
#define KK1 1000
#define FF 128
#define CC 256   // F*(DEPTH+1)
#define PP 64    // proxies

typedef __bf16 v16bf __attribute__((ext_vector_type(16)));
typedef float  v8f   __attribute__((ext_vector_type(8)));

__device__ __forceinline__ __bf16 f2bf(float x) {
    unsigned int u = __builtin_bit_cast(unsigned int, x);
    unsigned int r = (u + 0x7FFFu + ((u >> 16) & 1u)) >> 16;
    return __builtin_bit_cast(__bf16, (unsigned short)r);
}

__device__ __forceinline__ float waveAllReduce(float v) {
    #pragma unroll
    for (int o = 16; o > 0; o >>= 1) v += __shfl_xor(v, o, 32);
    return v;
}

// ---------------- Edge phase ----------------

// One wave32 per edge: logits[h] = (sv * rel_emb[rid]) . attn_kernels[h]
__global__ __launch_bounds__(256) void nrga_edge_logits(
    const float* __restrict__ rel_emb, const float* __restrict__ sparse_val,
    const int* __restrict__ sparse_idx, const float* __restrict__ attn,
    float* __restrict__ att)
{
    int wid  = (int)((blockIdx.x * blockDim.x + threadIdx.x) >> 5);
    int lane = threadIdx.x & 31;
    if (wid >= EE) return;
    int   rid = sparse_idx[2 * wid + 1];
    float sv  = sparse_val[wid];
    const float* rrow = rel_emb + (size_t)rid * FF;
    float d0 = 0.f, d1 = 0.f;
    #pragma unroll
    for (int i = 0; i < 4; ++i) {
        int f = i * 32 + lane;
        float r = sv * rrow[f];
        d0 += r * attn[f];
        d1 += r * attn[FF + f];
    }
    d0 = waveAllReduce(d0);
    d1 = waveAllReduce(d1);
    if (lane == 0) { att[2 * wid] = d0; att[2 * wid + 1] = d1; }
}

// rows in adj are sorted ascending -> CSR row_ptr
__global__ __launch_bounds__(256) void nrga_rowptr(
    const int* __restrict__ adj, int* __restrict__ row_ptr)
{
    int e = blockIdx.x * blockDim.x + threadIdx.x;
    if (e >= EE) return;
    int r1 = adj[2 * e];
    int r0 = (e == 0) ? -1 : adj[2 * (e - 1)];
    for (int r = r0 + 1; r <= r1; ++r) row_ptr[r] = e;
    if (e == EE - 1) for (int r = r1 + 1; r <= NN; ++r) row_ptr[r] = EE;
}

// thread-per-node segment softmax (both heads), normalizes att in place
__global__ __launch_bounds__(256) void nrga_seg_softmax(
    const int* __restrict__ row_ptr, float* __restrict__ att)
{
    int n = blockIdx.x * blockDim.x + threadIdx.x;
    if (n >= NN) return;
    int s = row_ptr[n], t = row_ptr[n + 1];
    #pragma unroll
    for (int h = 0; h < 2; ++h) {
        float mx = -1e30f;
        for (int e = s; e < t; ++e) mx = fmaxf(mx, att[2 * e + h]);
        float sum = 0.f;
        for (int e = s; e < t; ++e) { float v = __expf(att[2 * e + h] - mx); att[2 * e + h] = v; sum += v; }
        float inv = (sum > 0.f) ? 1.0f / sum : 0.0f;
        for (int e = s; e < t; ++e) att[2 * e + h] *= inv;
    }
}

// one wave32 per node: recompute rels_norm, reflect neighbor feats, accumulate heads
__global__ __launch_bounds__(256) void nrga_aggregate(
    const float* __restrict__ features, const float* __restrict__ rel_emb,
    const float* __restrict__ sparse_val, const int* __restrict__ adj,
    const int* __restrict__ sparse_idx, const int* __restrict__ row_ptr,
    const float* __restrict__ att,
    float* __restrict__ nf0, float* __restrict__ nf1)
{
    int wid  = (int)((blockIdx.x * blockDim.x + threadIdx.x) >> 5);
    int lane = threadIdx.x & 31;
    if (wid >= NN) return;
    int s = row_ptr[wid], t = row_ptr[wid + 1];
    float a0[4] = {0, 0, 0, 0}, a1[4] = {0, 0, 0, 0};
    for (int e = s; e < t; ++e) {
        int   col = adj[2 * e + 1];
        int   rid = sparse_idx[2 * e + 1];
        float sv  = sparse_val[e];
        float r[4], nb[4];
        float ss = 0.f, dd = 0.f;
        #pragma unroll
        for (int i = 0; i < 4; ++i) {
            int f = i * 32 + lane;
            r[i]  = sv * rel_emb[(size_t)rid * FF + f];
            ss   += r[i] * r[i];
            nb[i] = features[(size_t)col * FF + f];
        }
        ss = waveAllReduce(ss);
        float rn = rsqrtf(fmaxf(ss, 1e-12f));
        #pragma unroll
        for (int i = 0; i < 4; ++i) { r[i] *= rn; dd += nb[i] * r[i]; }
        dd = waveAllReduce(dd);
        float w0 = att[2 * e], w1 = att[2 * e + 1];
        #pragma unroll
        for (int i = 0; i < 4; ++i) {
            float refl = nb[i] - 2.0f * dd * r[i];
            a0[i] += refl * w0;
            a1[i] += refl * w1;
        }
    }
    #pragma unroll
    for (int i = 0; i < 4; ++i) {
        int f = i * 32 + lane;
        nf0[(size_t)wid * FF + f] = a0[i];
        nf1[(size_t)wid * FF + f] = a1[i];
    }
}

// feats1 = 0.5*(nf0 + head1-with-self-swap)  (nf0 lives in feats1 buffer)
__global__ __launch_bounds__(256) void nrga_combine(
    const int* __restrict__ neigh, const float* __restrict__ nf1,
    float* __restrict__ feats1)
{
    int i = blockIdx.x * blockDim.x + threadIdx.x;
    if (i >= NN * FF) return;
    int n = i >> 7, f = i & (FF - 1);
    int src = (n < KK1) ? neigh[n] : n;
    feats1[i] = 0.5f * (feats1[i] + nf1[(size_t)src * FF + f]);
}

// ---------------- Dense-phase prep ----------------

__global__ void nrga_pnorm(const float* __restrict__ proxy, float* __restrict__ pnorm) {
    int p = blockIdx.x * blockDim.x + threadIdx.x;
    if (p >= PP) return;
    float s = 0.f;
    for (int c = 0; c < CC; ++c) { float v = proxy[p * CC + c]; s += v * v; }
    pnorm[p] = rsqrtf(fmaxf(s, 1e-12f));
}

// Pack weights into WMMA B-fragment order (ISA 16-bit layout):
// fragment element j of lane L in tile (kt,nt) holds B[kt*32 + 16*(j/8) + 8*(L/16) + (j%8)][nt*16 + L%16]
__global__ __launch_bounds__(256) void nrga_pack_frags(
    const float* __restrict__ proxy, const float* __restrict__ gate,
    const float* __restrict__ pnorm,
    __bf16* __restrict__ fragPT, __bf16* __restrict__ fragP, __bf16* __restrict__ fragG)
{
    int idx = blockIdx.x * blockDim.x + threadIdx.x;
    // fragPT: B = l2norm(proxy)^T  (256 x 64): kt 0..7, nt 0..3 -> 16384 elems
    if (idx < 16384) {
        int j = idx & 15, lane = (idx >> 4) & 31, nt = (idx >> 9) & 3, kt = idx >> 11;
        int k = kt * 32 + ((j >= 8) ? 16 : 0) + ((lane >= 16) ? 8 : 0) + (j & 7);
        int n = nt * 16 + (lane & 15);
        fragPT[idx] = f2bf(proxy[n * CC + k] * pnorm[n]);
    }
    // fragP: B = proxy (64 x 256): kt 0..1, nt 0..15 -> 16384 elems
    if (idx < 16384) {
        int j = idx & 15, lane = (idx >> 4) & 31, nt = (idx >> 9) & 15, kt = idx >> 13;
        int k = kt * 32 + ((j >= 8) ? 16 : 0) + ((lane >= 16) ? 8 : 0) + (j & 7);
        int n = nt * 16 + (lane & 15);
        fragP[idx] = f2bf(proxy[k * CC + n]);
    }
    // fragG: B = gate_kernel (256 x 256): kt 0..7, nt 0..15 -> 65536 elems
    if (idx < 65536) {
        int j = idx & 15, lane = (idx >> 4) & 31, nt = (idx >> 9) & 15, kt = idx >> 13;
        int k = kt * 32 + ((j >= 8) ? 16 : 0) + ((lane >= 16) ? 8 : 0) + (j & 7);
        int n = nt * 16 + (lane & 15);
        fragG[idx] = f2bf(gate[k * CC + n]);
    }
}

// ---------------- Dense phase: WMMA tile kernel ----------------
// 1 block = 128 threads = 4 wave32s, handles 16 nodes.
__global__ __launch_bounds__(128) void nrga_dense(
    const float* __restrict__ features, const float* __restrict__ feats1,
    const __bf16* __restrict__ fragPT, const __bf16* __restrict__ fragP,
    const __bf16* __restrict__ fragG, float* __restrict__ out)
{
    __shared__ float  sOut[16][CC];
    __shared__ __bf16 sOutN[16][CC];
    __shared__ float  sLogit[16][PP];
    __shared__ __bf16 sAttB[16][PP];
    __shared__ float  sPf[16][CC];
    __shared__ __bf16 sPfB[16][CC];
    __shared__ float  sRinv[16];

    int tid = threadIdx.x;
    int node0 = blockIdx.x * 16;

    for (int i = tid; i < 16 * CC; i += 128) {
        int m = i >> 8, c = i & (CC - 1);
        float v = (c < FF) ? features[(size_t)(node0 + m) * FF + c]
                           : feats1[(size_t)(node0 + m) * FF + (c - FF)];
        sOut[m][c] = v;
    }
    __syncthreads();
    if (tid < 16) {
        float s = 0.f;
        for (int c = 0; c < CC; ++c) { float v = sOut[tid][c]; s += v * v; }
        sRinv[tid] = rsqrtf(fmaxf(s, 1e-12f));
    }
    __syncthreads();
    for (int i = tid; i < 16 * CC; i += 128) {
        int m = i >> 8, c = i & (CC - 1);
        sOutN[m][c] = f2bf(sOut[m][c] * sRinv[m]);
    }
    __syncthreads();

    int lane = tid & 31, w = tid >> 5;
    int mrow = lane & 15;
    int khi  = (lane >> 4) << 3;   // 0 or 8
    int mhi  = (lane >> 4) << 3;   // C/D row offset

    // GEMM1: logits(16x64) = sOutN(16x256) @ proxyN^T ; wave w owns nt = w
    {
        v8f acc = {};
        for (int kt = 0; kt < 8; ++kt) {
            v16bf a, b;
            int kb = kt * 32 + khi;
            #pragma unroll
            for (int j = 0; j < 16; ++j)
                a[j] = sOutN[mrow][kb + ((j >= 8) ? 16 : 0) + (j & 7)];
            b = ((const v16bf*)fragPT)[(kt * 4 + w) * 32 + lane];
            acc = __builtin_amdgcn_wmma_f32_16x16x32_bf16(false, a, false, b, (short)0, acc, false, false);
        }
        #pragma unroll
        for (int v = 0; v < 8; ++v)
            sLogit[v + mhi][w * 16 + (lane & 15)] = acc[v];
    }
    __syncthreads();

    // softmax over 64 proxies per node
    if (tid < 16) {
        float mx = -1e30f;
        for (int p = 0; p < PP; ++p) mx = fmaxf(mx, sLogit[tid][p]);
        float s = 0.f;
        for (int p = 0; p < PP; ++p) { float e = __expf(sLogit[tid][p] - mx); sLogit[tid][p] = e; s += e; }
        float inv = 1.0f / s;
        for (int p = 0; p < PP; ++p) sAttB[tid][p] = f2bf(sLogit[tid][p] * inv);
    }
    __syncthreads();

    // GEMM2: pf = out - att(16x64) @ proxy(64x256); wave owns nt = 4w..4w+3
    for (int t = 0; t < 4; ++t) {
        int nt = w * 4 + t;
        v8f acc = {};
        for (int kt = 0; kt < 2; ++kt) {
            v16bf a, b;
            int kb = kt * 32 + khi;
            #pragma unroll
            for (int j = 0; j < 16; ++j)
                a[j] = sAttB[mrow][kb + ((j >= 8) ? 16 : 0) + (j & 7)];
            b = ((const v16bf*)fragP)[(kt * 16 + nt) * 32 + lane];
            acc = __builtin_amdgcn_wmma_f32_16x16x32_bf16(false, a, false, b, (short)0, acc, false, false);
        }
        #pragma unroll
        for (int v = 0; v < 8; ++v) {
            int m = v + mhi, gn = nt * 16 + (lane & 15);
            float pf = sOut[m][gn] - acc[v];
            sPf[m][gn]  = pf;
            sPfB[m][gn] = f2bf(pf);
        }
    }
    __syncthreads();

    // GEMM3: gate = sigmoid(pf @ gate_kernel); blend and store
    for (int t = 0; t < 4; ++t) {
        int nt = w * 4 + t;
        v8f acc = {};
        for (int kt = 0; kt < 8; ++kt) {
            v16bf a, b;
            int kb = kt * 32 + khi;
            #pragma unroll
            for (int j = 0; j < 16; ++j)
                a[j] = sPfB[mrow][kb + ((j >= 8) ? 16 : 0) + (j & 7)];
            b = ((const v16bf*)fragG)[(kt * 16 + nt) * 32 + lane];
            acc = __builtin_amdgcn_wmma_f32_16x16x32_bf16(false, a, false, b, (short)0, acc, false, false);
        }
        #pragma unroll
        for (int v = 0; v < 8; ++v) {
            int m = v + mhi, gn = nt * 16 + (lane & 15);
            float g = 1.0f / (1.0f + __expf(-acc[v]));
            float o = sOut[m][gn], p = sPf[m][gn];
            out[(size_t)(node0 + m) * CC + gn] = g * o + (1.0f - g) * p;
        }
    }
}

// ---------------- Launch ----------------

extern "C" void kernel_launch(void* const* d_in, const int* in_sizes, int n_in,
                              void* d_out, int out_size, void* d_ws, size_t ws_size,
                              hipStream_t stream) {
    (void)in_sizes; (void)n_in; (void)out_size; (void)ws_size;

    const float* features   = (const float*)d_in[0];
    const float* rel_emb    = (const float*)d_in[1];
    const float* sparse_val = (const float*)d_in[2];
    const float* attn       = (const float*)d_in[3];
    const float* proxy      = (const float*)d_in[4];
    const float* gate       = (const float*)d_in[5];
    const int*   adj        = (const int*)d_in[6];
    const int*   sparse_idx = (const int*)d_in[7];
    const int*   neigh      = (const int*)d_in[9];
    float*       out        = (float*)d_out;

    char* ws = (char*)d_ws;
    size_t off = 0;
    auto alloc = [&](size_t bytes) -> void* {
        void* p = ws + off;
        off = (off + bytes + 255) & ~(size_t)255;
        return p;
    };
    float*  att     = (float*)alloc((size_t)EE * 2 * sizeof(float));
    int*    row_ptr = (int*)  alloc((size_t)(NN + 1) * sizeof(int));
    float*  feats1  = (float*)alloc((size_t)NN * FF * sizeof(float)); // nf0 -> layer output
    float*  nf1     = (float*)alloc((size_t)NN * FF * sizeof(float));
    float*  pnorm   = (float*)alloc(PP * sizeof(float));
    __bf16* fragPT  = (__bf16*)alloc(16384 * sizeof(__bf16));
    __bf16* fragP   = (__bf16*)alloc(16384 * sizeof(__bf16));
    __bf16* fragG   = (__bf16*)alloc(65536 * sizeof(__bf16));

    nrga_edge_logits<<<EE / 8, 256, 0, stream>>>(rel_emb, sparse_val, sparse_idx, attn, att);
    nrga_rowptr<<<(EE + 255) / 256, 256, 0, stream>>>(adj, row_ptr);
    nrga_seg_softmax<<<(NN + 255) / 256, 256, 0, stream>>>(row_ptr, att);
    nrga_aggregate<<<(NN + 7) / 8, 256, 0, stream>>>(features, rel_emb, sparse_val, adj,
                                                     sparse_idx, row_ptr, att, feats1, nf1);
    nrga_combine<<<(NN * FF + 255) / 256, 256, 0, stream>>>(neigh, nf1, feats1);
    nrga_pnorm<<<1, 64, 0, stream>>>(proxy, pnorm);
    nrga_pack_frags<<<256, 256, 0, stream>>>(proxy, gate, pnorm, fragPT, fragP, fragG);
    nrga_dense<<<NN / 16, 128, 0, stream>>>(features, feats1, fragPT, fragP, fragG, out);
}
